// DeformationGNN_18863496364541
// MI455X (gfx1250) — matrix operand order repaired and checked by reading, not compiled, not run on hardware
//
#include <hip/hip_runtime.h>
#include <hip/hip_bf16.h>

typedef __attribute__((ext_vector_type(16))) __bf16 v16bf;
typedef __attribute__((ext_vector_type(8)))  float  v8f;

#define GN   100000
#define GE   1200000
#define HIDD 64

// ---------------- norm precompute ----------------

__global__ __launch_bounds__(256) void deg_init(float* deg) {
    int i = blockIdx.x * blockDim.x + threadIdx.x;
    if (i < GN) deg[i] = 1.0f;                 // self-loop weight
}

__global__ __launch_bounds__(256) void deg_accum(const long long* __restrict__ ei,
                                                 const float* __restrict__ w,
                                                 float* deg) {
    int e = blockIdx.x * blockDim.x + threadIdx.x;
    if (e >= GE) return;
    int c = (int)ei[GE + e];                   // col = target
    atomicAdd(&deg[c], w[e]);
}

__global__ __launch_bounds__(256) void deg_rsqrt(float* deg) {
    int i = blockIdx.x * blockDim.x + threadIdx.x;
    if (i < GN) deg[i] = rsqrtf(deg[i]);       // deg >= 1 always
}

__global__ __launch_bounds__(256) void norm_kernel(const long long* __restrict__ ei,
                                                   const float* __restrict__ w,
                                                   const float* __restrict__ dinv,
                                                   float* __restrict__ nrm) {
    int e = blockIdx.x * blockDim.x + threadIdx.x;
    if (e >= GE) return;
    int r = (int)ei[e];
    int c = (int)ei[GE + e];
    nrm[e] = dinv[r] * w[e] * dinv[c];
}

// ---------------- WMMA GEMMs ----------------
// One wave -> one 16-row node tile, loops 4 col-tiles of the 64-wide output.

__global__ __launch_bounds__(256) void gemm_xw1(const float* __restrict__ x,
                                                const float* __restrict__ W,
                                                float* __restrict__ out) {
    const int wid = (blockIdx.x * blockDim.x + threadIdx.x) >> 5;
    if (wid >= GN / 16) return;                // wave-uniform exit
    const int lane = threadIdx.x & 31;
    const int lrow = lane & 15;
    const int hi   = lane >> 4;

    const float* xr = x + (size_t)(wid * 16 + lrow) * 32;
    v16bf a;
#pragma unroll
    for (int j = 0; j < 8; ++j) {              // 16-bit A 16x32 lane layout
        const int k = ((j < 4) ? (2 * j) : (16 + 2 * (j - 4))) + hi * 8;
        a[2 * j]     = (__bf16)xr[k];
        a[2 * j + 1] = (__bf16)xr[k + 1];
    }
#pragma unroll
    for (int nt = 0; nt < 4; ++nt) {
        const int col = nt * 16 + lrow;
        v16bf b;
#pragma unroll
        for (int j = 0; j < 8; ++j) {          // 16-bit B 32x16 lane layout
            const int k0 = ((j < 4) ? (2 * j) : (16 + 2 * (j - 4))) + hi * 8;
            b[2 * j]     = (__bf16)W[k0 * HIDD + col];
            b[2 * j + 1] = (__bf16)W[(k0 + 1) * HIDD + col];
        }
        v8f c = {};
        c = __builtin_amdgcn_wmma_f32_16x16x32_bf16(false, a, false, b,
                                                    (short)0, c, false, false);
#pragma unroll
        for (int j = 0; j < 8; ++j)
            out[(size_t)(wid * 16 + hi * 8 + j) * HIDD + col] = c[j];
    }
}

__global__ __launch_bounds__(256) void gemm_hw2(const __bf16* __restrict__ h,
                                                const float* __restrict__ W,
                                                float* __restrict__ out) {
    const int wid = (blockIdx.x * blockDim.x + threadIdx.x) >> 5;
    if (wid >= GN / 16) return;
    const int lane = threadIdx.x & 31;
    const int lrow = lane & 15;
    const int hi   = lane >> 4;

    const __bf16* hr = h + (size_t)(wid * 16 + lrow) * HIDD;
    v16bf a0, a1;                              // K = 0..31 and K = 32..63
#pragma unroll
    for (int j = 0; j < 8; ++j) {
        const int k = ((j < 4) ? (2 * j) : (16 + 2 * (j - 4))) + hi * 8;
        a0[2 * j]     = hr[k];
        a0[2 * j + 1] = hr[k + 1];
        a1[2 * j]     = hr[32 + k];
        a1[2 * j + 1] = hr[32 + k + 1];
    }
#pragma unroll
    for (int nt = 0; nt < 4; ++nt) {
        const int col = nt * 16 + lrow;
        v16bf b0, b1;
#pragma unroll
        for (int j = 0; j < 8; ++j) {
            const int k0 = ((j < 4) ? (2 * j) : (16 + 2 * (j - 4))) + hi * 8;
            b0[2 * j]     = (__bf16)W[k0 * HIDD + col];
            b0[2 * j + 1] = (__bf16)W[(k0 + 1) * HIDD + col];
            b1[2 * j]     = (__bf16)W[(32 + k0) * HIDD + col];
            b1[2 * j + 1] = (__bf16)W[(33 + k0) * HIDD + col];
        }
        v8f c = {};
        c = __builtin_amdgcn_wmma_f32_16x16x32_bf16(false, a0, false, b0,
                                                    (short)0, c, false, false);
        c = __builtin_amdgcn_wmma_f32_16x16x32_bf16(false, a1, false, b1,
                                                    (short)0, c, false, false);
#pragma unroll
        for (int j = 0; j < 8; ++j)
            out[(size_t)(wid * 16 + hi * 8 + j) * HIDD + col] = c[j];
    }
}

// ---------------- message passing ----------------

// agg[i] = dinv[i]^2 * feat[i]   (self-loop contribution; full init of agg)
__global__ __launch_bounds__(256) void self_init(const float* __restrict__ feat,
                                                 const float* __restrict__ dinv,
                                                 float* __restrict__ agg) {
    int g = blockIdx.x * blockDim.x + threadIdx.x;
    if (g >= GN * HIDD) return;
    int i = g >> 6;
    float d = dinv[i];
    agg[g] = d * d * feat[g];
}

// per (edge, 4-channel chunk): gather float4, scale by norm, scatter-atomic-add
__global__ __launch_bounds__(256) void edge_scatter(const long long* __restrict__ ei,
                                                    const float* __restrict__ nrm,
                                                    const float* __restrict__ feat,
                                                    float* __restrict__ agg) {
    long long g = (long long)blockIdx.x * blockDim.x + threadIdx.x;
    int e = (int)(g >> 4);
    if (e >= GE) return;
    int c4 = ((int)g & 15) * 4;
    int r = (int)ei[e];
    int c = (int)ei[GE + e];
    float nm = nrm[e];
    const float4 v = *(const float4*)(feat + (size_t)r * HIDD + c4);
    float* dst = agg + (size_t)c * HIDD + c4;
    atomicAdd(dst + 0, nm * v.x);
    atomicAdd(dst + 1, nm * v.y);
    atomicAdd(dst + 2, nm * v.z);
    atomicAdd(dst + 3, nm * v.w);
}

__global__ __launch_bounds__(256) void bias_relu(const float* __restrict__ agg,
                                                 const float* __restrict__ bias,
                                                 float* __restrict__ h32,
                                                 __bf16* __restrict__ hbf) {
    int g = blockIdx.x * blockDim.x + threadIdx.x;
    if (g >= GN * HIDD) return;
    float v = fmaxf(agg[g] + bias[g & (HIDD - 1)], 0.0f);
    h32[g] = v;
    hbf[g] = (__bf16)v;
}

// out[i] = h2[i] @ Wf + bf   (64x3, trivial)
__global__ __launch_bounds__(256) void final_proj(const float* __restrict__ h,
                                                  const float* __restrict__ Wf,
                                                  const float* __restrict__ bfv,
                                                  float* __restrict__ out) {
    int i = blockIdx.x * blockDim.x + threadIdx.x;
    if (i >= GN) return;
    const float* hr = h + (size_t)i * HIDD;
    float a0 = bfv[0], a1 = bfv[1], a2 = bfv[2];
#pragma unroll
    for (int c = 0; c < HIDD; ++c) {
        float v = hr[c];
        a0 += v * Wf[c * 3 + 0];
        a1 += v * Wf[c * 3 + 1];
        a2 += v * Wf[c * 3 + 2];
    }
    out[i * 3 + 0] = a0;
    out[i * 3 + 1] = a1;
    out[i * 3 + 2] = a2;
}

// ---------------- host launch ----------------

static inline size_t alignup(size_t v) { return (v + 255) & ~(size_t)255; }

extern "C" void kernel_launch(void* const* d_in, const int* in_sizes, int n_in,
                              void* d_out, int out_size, void* d_ws, size_t ws_size,
                              hipStream_t stream) {
    (void)in_sizes; (void)n_in; (void)out_size; (void)ws_size;
    const float*      x  = (const float*)d_in[0];
    const long long*  ei = (const long long*)d_in[1];   // int64 (2, E)
    const float*      ew = (const float*)d_in[2];
    const float*      W1 = (const float*)d_in[3];
    const float*      b1 = (const float*)d_in[4];
    const float*      W2 = (const float*)d_in[5];
    const float*      b2 = (const float*)d_in[6];
    const float*      Wf = (const float*)d_in[7];
    const float*      bf = (const float*)d_in[8];
    float*            out = (float*)d_out;

    char* p = (char*)d_ws;
    float*  dinv = (float*)p;  p += alignup((size_t)GN * 4);
    float*  nrm  = (float*)p;  p += alignup((size_t)GE * 4);
    float*  bufA = (float*)p;  p += alignup((size_t)GN * HIDD * 4);
    float*  bufB = (float*)p;  p += alignup((size_t)GN * HIDD * 4);
    __bf16* bufH = (__bf16*)p; p += alignup((size_t)GN * HIDD * 2);

    const int nodeB  = (GN + 255) / 256;
    const int edgeB  = (GE + 255) / 256;
    const int featB  = (GN * HIDD + 255) / 256;
    const int scatB  = (int)(((long long)GE * 16 + 255) / 256);
    const int gemmB  = (GN / 16 + 7) / 8;      // 8 waves per 256-thread block

    // norm precompute
    deg_init   <<<nodeB, 256, 0, stream>>>(dinv);
    deg_accum  <<<edgeB, 256, 0, stream>>>(ei, ew, dinv);
    deg_rsqrt  <<<nodeB, 256, 0, stream>>>(dinv);
    norm_kernel<<<edgeB, 256, 0, stream>>>(ei, ew, dinv, nrm);

    // layer 1: x@W1 -> aggregate -> relu(+b1)
    gemm_xw1    <<<gemmB, 256, 0, stream>>>(x, W1, bufA);
    self_init   <<<featB, 256, 0, stream>>>(bufA, dinv, bufB);
    edge_scatter<<<scatB, 256, 0, stream>>>(ei, nrm, bufA, bufB);
    bias_relu   <<<featB, 256, 0, stream>>>(bufB, b1, bufA, bufH);

    // layer 2: h1@W2 -> aggregate -> relu(+b2)
    gemm_hw2    <<<gemmB, 256, 0, stream>>>(bufH, W2, bufA);
    self_init   <<<featB, 256, 0, stream>>>(bufA, dinv, bufB);
    edge_scatter<<<scatB, 256, 0, stream>>>(ei, nrm, bufA, bufB);
    bias_relu   <<<featB, 256, 0, stream>>>(bufB, b2, bufA, bufH);

    // final projection h2 @ Wf + bf
    final_proj  <<<nodeB, 256, 0, stream>>>(bufA, Wf, bf, out);
}
